// QuantLayer_20298015441556
// MI455X (gfx1250) — compile-verified, matrix operand order checked by reading, not run on hardware
//
#include <hip/hip_runtime.h>

// ---------------------------------------------------------------------------
// Binary-weight 3x3 conv as implicit GEMM on CDNA5 WMMA (gfx1250, wave32).
//
// out[b,co,h,w] = scale[co] * sum_{ci,kh,kw} sign_w[co,ci,kh,kw]*x[b,ci,h+kh-1,w+kw-1]
//              + bias[co]
//
// Roofline: 1.18e11 FLOP vs ~208 MB HBM traffic -> AI ~568 FLOP/B; at 23.3 TB/s
// the memory floor is ~9us => matrix-compute bound. Use v_wmma_f32_16x16x32_f16
// (sign weights exact in f16, activations ~N(0,1) fit f16), f32 accumulation.
//
// CDNA5 data movement:
//  - A (weight) tile: Tensor Data Mover (tensor_load_to_lds, 6-arg clang-23
//    builtin), one descriptor per K-step issued by wave 0, synced with
//    s_wait_tensorcnt + workgroup barrier.
//  - B (activation) tile: per-lane gather + f32->f16 convert -> ds_store.
// ---------------------------------------------------------------------------

typedef __attribute__((ext_vector_type(16))) _Float16 v16h;
typedef __attribute__((ext_vector_type(8)))  _Float16 v8h;
typedef __attribute__((ext_vector_type(8)))  float    v8f;
typedef __attribute__((ext_vector_type(4)))  unsigned u32x4;
typedef __attribute__((ext_vector_type(8)))  int      i32x8;
typedef __attribute__((ext_vector_type(4)))  int      i32x4;

#define CIN   256
#define COUT  256
#define HWDIM 56
#define IMG   (HWDIM * HWDIM)     // 3136
#define NB    32
#define NPIX  (NB * IMG)          // 100352
#define BM    128                 // Cout tile
#define BN    128                 // pixel tile
#define BK    32                  // K step (f16 WMMA K)
#define KSTEPS (9 * (CIN / BK))   // 72

union V16U { v16h v; v8h h[2]; };

// --- Prologue: pack sign weights [Cout][Cin][3][3] f32 -> [kpos][Cout][Cin] f16
__global__ void pack_weights(const float* __restrict__ qw,
                             _Float16* __restrict__ wpk) {
    int idx = blockIdx.x * 256 + threadIdx.x;     // over COUT*CIN*9
    if (idx >= COUT * CIN * 9) return;
    int kw = idx % 3;
    int t  = idx / 3;
    int kh = t % 3;
    t /= 3;
    int ci = t % CIN;
    int co = t / CIN;
    int kpos = kh * 3 + kw;
    wpk[((long)kpos * COUT + co) * CIN + ci] = (_Float16)qw[idx];
}

// --- Main implicit-GEMM kernel -------------------------------------------
__global__ void __launch_bounds__(256)
binconv_wmma(const float* __restrict__ x,
             const _Float16* __restrict__ wpk,
             const float* __restrict__ scale,
             const float* __restrict__ bias,
             float* __restrict__ out) {
    // Double-buffered LDS tiles. A: [m][k] row-major (matches TDM tile layout:
    // 128 rows x 64B contiguous). B: [n][k] so each lane's WMMA fragment is two
    // contiguous 16B chunks.
    __shared__ __attribute__((aligned(16))) _Float16 As[2][BM * BK];
    __shared__ __attribute__((aligned(16))) _Float16 Bs[2][BN * BK];

    const int tid   = threadIdx.x;
    const int lane  = tid & 31;
    const int wid   = tid >> 5;
    const int waveM = wid >> 2;            // 0..1  -> 64 rows each
    const int waveN = wid & 3;             // 0..3  -> 32 cols each
    const int m0    = blockIdx.y * BM;
    const int n0    = blockIdx.x * BN;

    // B-staging: thread handles one pixel column, 16 of the 32 K rows.
    const int  bc   = tid & 127;           // pixel column within tile
    const int  kh2  = tid >> 7;            // 0/1 -> K rows [kh2*16, kh2*16+16)
    const int  pix  = n0 + bc;
    const int  pb   = pix / IMG;
    const int  prem = pix - pb * IMG;
    const int  ph   = prem / HWDIM;
    const int  pw   = prem - ph * HWDIM;
    const long xbase = (long)pb * CIN * IMG;

    const v8f vzero = {};
    v8f acc[4][2];
#pragma unroll
    for (int mt = 0; mt < 4; ++mt)
#pragma unroll
        for (int nt = 0; nt < 2; ++nt)
            acc[mt][nt] = vzero;

    auto stage = [&](int ks, int buf) {
        const int kpos = ks >> 3;          // kernel tap 0..8
        const int k0   = (ks & 7) * BK;    // Cin offset
        const int dh   = kpos / 3 - 1;
        const int dw   = kpos % 3 - 1;

        // ---- A tile via Tensor Data Mover: 128 x 32 f16 weights ----
        // 2-D D#: tensor = [2304 rows x 256 cols] f16, tile = [128 x 32] at
        // (row = kpos*256 + m0, col = k0). Tile rows land contiguously in LDS.
        {
            const unsigned long long gaddr =
                (unsigned long long)(const void*)(wpk + ((long)kpos * COUT + m0) * CIN + k0);
            const unsigned ldsoff =
                (unsigned)(unsigned long long)(const void*)&As[buf][0];
            u32x4 g0;
            g0.x = 1u;                                    // count=1, user desc
            g0.y = ldsoff;                                // lds_addr (bytes)
            g0.z = (unsigned)gaddr;                       // global_addr[31:0]
            g0.w = (unsigned)((gaddr >> 32) & 0x1FFFFFFu) // global_addr[56:32]
                   | (2u << 30);                          // type=2 (image)
            i32x8 g1;
            g1[0] = 0x00010000;                 // data_size=1 (2B), no multicast
            g1[1] = (int)(256u << 16);          // tensor_dim0 = 256 (low16)
            g1[2] = (int)((9u * COUT) << 16);   // dim0 hi=0 | tensor_dim1 lo=2304
            g1[3] = (int)(32u << 16);           // dim1 hi=0 | tile_dim0 = 32
            g1[4] = 128;                        // tile_dim1=128 | tile_dim2=0
            g1[5] = 256;                        // tensor_dim0_stride lo32 = 256
            g1[6] = 0;
            g1[7] = 0;
            const i32x4 gz4 = {0, 0, 0, 0};
            const i32x8 gz8 = {0, 0, 0, 0, 0, 0, 0, 0};
            if (wid == 0)
                __builtin_amdgcn_tensor_load_to_lds(g0, g1, gz4, gz4, gz8, 0);
        }

        // ---- B tile: 32 x 128 f16 shifted activations (gather + cvt) ----
        const int hh = ph + dh;
        const int ww = pw + dw;
        const bool valid = (hh >= 0) & (hh < HWDIM) & (ww >= 0) & (ww < HWDIM);
        const float* xsrc = x + xbase + (long)(k0 + kh2 * 16) * IMG
                              + hh * HWDIM + ww;
        _Float16* bdst = &Bs[buf][bc * BK + kh2 * 16];
        // prefetch next K-step's activations (emits global_prefetch_b8)
        __builtin_prefetch(xsrc + (long)BK * IMG, 0, 0);
#pragma unroll
        for (int j = 0; j < 16; ++j) {
            float v = valid ? xsrc[(long)j * IMG] : 0.0f;
            bdst[j] = (_Float16)v;
        }
    };

    auto compute = [&](int buf) {
        const int lrow = lane & 15;
        const int ko   = (lane >> 4) * 8;  // per-ISA 16-bit A/B lane K split
        V16U a[4], b[2];
#pragma unroll
        for (int mt = 0; mt < 4; ++mt) {
            const _Float16* p = &As[buf][(waveM * 64 + mt * 16 + lrow) * BK + ko];
            a[mt].h[0] = *(const v8h*)p;          // K = ko .. ko+7
            a[mt].h[1] = *(const v8h*)(p + 16);   // K = ko+16 .. ko+23
        }
#pragma unroll
        for (int nt = 0; nt < 2; ++nt) {
            const _Float16* p = &Bs[buf][(waveN * 32 + nt * 16 + lrow) * BK + ko];
            b[nt].h[0] = *(const v8h*)p;
            b[nt].h[1] = *(const v8h*)(p + 16);
        }
#pragma unroll
        for (int mt = 0; mt < 4; ++mt)
#pragma unroll
            for (int nt = 0; nt < 2; ++nt)
                acc[mt][nt] = __builtin_amdgcn_wmma_f32_16x16x32_f16(
                    false, a[mt].v, false, b[nt].v,
                    (short)0, acc[mt][nt], false, false);
    };

    // Software pipeline: stage(ks+1) (TDM + B gather) overlaps compute(ks).
    // s_wait_tensorcnt(0) retires wave 0's TDM before the workgroup barrier
    // publishes the freshly staged buffer.
    stage(0, 0);
    __builtin_amdgcn_s_wait_tensorcnt(0);
    __syncthreads();
    for (int ks = 0; ks < KSTEPS; ++ks) {
        if (ks + 1 < KSTEPS) stage(ks + 1, (ks + 1) & 1);
        compute(ks & 1);
        __builtin_amdgcn_s_wait_tensorcnt(0);
        __syncthreads();
    }

    // Epilogue: D layout = lane (N), VGPR v (M row), lanes>=16 are M+8.
    const int lcol = lane & 15;
    const int lro8 = (lane >> 4) * 8;
#pragma unroll
    for (int nt = 0; nt < 2; ++nt) {
        const int pixn = n0 + waveN * 32 + nt * 16 + lcol;
        const int ob   = pixn / IMG;
        const int ohw  = pixn - ob * IMG;
#pragma unroll
        for (int mt = 0; mt < 4; ++mt) {
#pragma unroll
            for (int v = 0; v < 8; ++v) {
                const int co = m0 + waveM * 64 + mt * 16 + lro8 + v;
                float r = acc[mt][nt][v] * scale[co] + bias[co];
                out[((long)ob * COUT + co) * IMG + ohw] = r;
            }
        }
    }
}

// ---------------------------------------------------------------------------
extern "C" void kernel_launch(void* const* d_in, const int* in_sizes, int n_in,
                              void* d_out, int out_size, void* d_ws, size_t ws_size,
                              hipStream_t stream) {
    const float* x     = (const float*)d_in[0];   // [32,256,56,56]
    const float* scale = (const float*)d_in[1];   // [256,1,1,1]
    const float* qw    = (const float*)d_in[2];   // [256,256,3,3] in {-1,0,1}
    const float* bias  = (const float*)d_in[3];   // [256]

    _Float16* wpk = (_Float16*)d_ws;              // 9*256*256 f16 = 1.18 MB

    pack_weights<<<(COUT * CIN * 9 + 255) / 256, 256, 0, stream>>>(qw, wpk);

    dim3 grid(NPIX / BN, COUT / BM);              // 784 x 2 blocks
    binconv_wmma<<<grid, 256, 0, stream>>>(x, wpk, scale, bias, (float*)d_out);
}
